// QJambaMambaMixer_24524263260386
// MI455X (gfx1250) — compile-verified, matrix operand order checked by reading, not run on hardware
//
#include <hip/hip_runtime.h>
#include <hip/hip_bf16.h>
#include <stdint.h>

// ---------------- constants from the reference ----------------
#define S_IN_f   0.05f
#define S_Z_f    0.02f
#define S_U_f    0.01f
#define S_DTLN_f 0.02f
#define S_B_f    0.02f
#define S_C_f    0.02f
#define S_DT_f   0.01f
#define S_Y_f    0.01f
#define EPS_f    1e-6f

typedef __attribute__((ext_vector_type(8))) int v8i;

static __device__ __forceinline__ float sigmoidf_(float x) { return 1.f / (1.f + __expf(-x)); }
static __device__ __forceinline__ float siluf_(float x)    { return x * sigmoidf_(x); }
static __device__ __forceinline__ float softplusf_(float x){ return (x > 20.f) ? x : log1pf(__expf(x)); }

// qa(x, s) = clip(round(x/s), -128, 127); rintf = round-half-even == jnp.round
static __device__ __forceinline__ int8_t qa8(float x, float inv_s) {
    float r = rintf(x * inv_s);
    r = fminf(fmaxf(r, -128.f), 127.f);
    return (int8_t)(int)r;
}

// CDNA5 async copy: 16B global -> LDS, tracked by ASYNCcnt (no VGPR round-trip).
static __device__ __forceinline__ void async_g2l_b128(const void* gsrc, uint32_t lds_addr) {
    asm volatile("global_load_async_to_lds_b128 %0, %1, off"
                 :: "v"(lds_addr), "v"(gsrc) : "memory");
}
static __device__ __forceinline__ void wait_asynccnt0() {
    asm volatile("s_wait_asynccnt 0x0" ::: "memory");
}

// ---------------- scale / quantize kernels ----------------
__global__ void k_zero_u32(unsigned* p, int n) {
    int i = blockIdx.x * blockDim.x + threadIdx.x;
    if (i < n) p[i] = 0u;
}

__global__ void k_absmax(const float* __restrict__ w, long n, unsigned* __restrict__ slot) {
    float m = 0.f;
    for (long i = (long)blockIdx.x * blockDim.x + threadIdx.x; i < n;
         i += (long)gridDim.x * blockDim.x)
        m = fmaxf(m, fabsf(w[i]));
    for (int off = 16; off; off >>= 1) m = fmaxf(m, __shfl_xor(m, off, 32)); // wave32
    if ((threadIdx.x & 31) == 0) atomicMax(slot, __float_as_uint(m));
}

__global__ void k_quant_w(const float* __restrict__ w, long n,
                          const unsigned* __restrict__ slot, int8_t* __restrict__ out) {
    long i = (long)blockIdx.x * blockDim.x + threadIdx.x;
    if (i >= n) return;
    float s = __uint_as_float(*slot) * (1.f / 127.f);
    float inv = (s > 0.f) ? (1.f / s) : 0.f;
    out[i] = qa8(w[i], inv);
}

__global__ void k_quant_x(const float* __restrict__ x, long n, float inv_s,
                          int8_t* __restrict__ out) {
    long i = (long)blockIdx.x * blockDim.x + threadIdx.x;
    if (i >= n) return;
    out[i] = qa8(x[i], inv_s);
}

// ---------------- WMMA int8 GEMM: C = A(MxK) . B(NxK)^T ----------------
// 8-bit 16x64 fragment per the CDNA5 ISA layout (lane = row/col index 0..15,
// half-wave selects which 8-byte sub-chunk of each 16-byte K group).
static __device__ __forceinline__ v8i load_frag_i8(const int8_t* __restrict__ base, int half) {
    const int* p = (const int*)base;   // 4-byte aligned
    const int o = half * 2;
    v8i f;
    f[0] = p[o + 0];  f[1] = p[o + 1];        // K = j*16 + half*8 .. +7, j=0
    f[2] = p[4 + o];  f[3] = p[4 + o + 1];    // j=1
    f[4] = p[8 + o];  f[5] = p[8 + o + 1];    // j=2
    f[6] = p[12 + o]; f[7] = p[12 + o + 1];   // j=3
    return f;
}
static __device__ __forceinline__ v8i load_frag_lds(const int8_t* base, int half) {
    const int* p = (const int*)base;   // addrspace(3) pointer -> ds_load
    const int o = half * 2;
    v8i f;
    f[0] = p[o + 0];  f[1] = p[o + 1];
    f[2] = p[4 + o];  f[3] = p[4 + o + 1];
    f[4] = p[8 + o];  f[5] = p[8 + o + 1];
    f[6] = p[12 + o]; f[7] = p[12 + o + 1];
    return f;
}

// One 64-wide K step: wait for this buffer's DMA, kick next slab's DMA into the
// other buffer, then interleave {B-fragment ds-load j -> WMMA j}: each fragment
// dies immediately (accumulators stay pinned) while WMMA j overlaps loads j+1.
template <int JT>
static __device__ __forceinline__ void gemm_step(
    const int8_t* __restrict__ arow, int k0, int K,
    const int8_t* bbase, bool do_copy, const int8_t* __restrict__ gsrc,
    uint32_t lds_next, int half, int l16, v8i (&acc)[JT]) {
    wait_asynccnt0();                 // my slab-copy into bbase done
    __syncthreads();                  // everyone's copy done; prior reads of next buf done
    if (do_copy && (k0 + 64 < K))
        async_g2l_b128(gsrc + k0 + 64, lds_next);   // overlap next slab DMA
    if (k0 + 64 < K) __builtin_prefetch(arow + k0 + 64, 0, 0);

    v8i af = load_frag_i8(arow + k0, half);
#pragma unroll
    for (int j = 0; j < JT; ++j) {
        v8i bf = load_frag_lds(bbase + (j * 16 + l16) * 64, half);
        acc[j] = __builtin_amdgcn_wmma_i32_16x16x64_iu8(
            true, af, true, bf, acc[j], false, false);   // signed x signed
    }
}

// Block: 256 threads (8 waves). Each wave owns 16 M-rows; block covers 128(M) x JT*16(N).
// B slab (JT*16 rows x 64 K-bytes) is async-DMA'd into double-buffered LDS; K loop is
// unrolled x2 so the buffer choice is compile-time. K % 128 == 0 for all launches.
// MODE 0: Out float = acc*scale ; MODE 1: Out int8 = qa(acc*scale, inv_qdiv)
template <int MODE, int JT>
__global__ void __launch_bounds__(256)
k_gemm_i8(const int8_t* __restrict__ A, const int8_t* __restrict__ B,
          int M, int N, int K, int nb0, void* __restrict__ Out,
          const unsigned* __restrict__ wmax_slot, float prefac, float inv_qdiv) {
    __shared__ __align__(16) int8_t ldsB[2][JT * 16 * 64];

    const int tid  = threadIdx.x;
    const int lane = tid & 31;
    const int wave = tid >> 5;
    const int half = lane >> 4;
    const int l16  = lane & 15;
    const int m0 = blockIdx.y * 128 + wave * 16;   // exact: M % 128 == 0 for all launches
    const int nb = nb0 + blockIdx.x * 64;

    v8i acc[JT];
#pragma unroll
    for (int j = 0; j < JT; ++j)
#pragma unroll
        for (int e = 0; e < 8; ++e) acc[j][e] = 0;

    // staging role: thread -> (row, 16B chunk) of the B slab
    const int  rr = tid >> 2;            // 0..63
    const int  cc = tid & 3;             // 0..3
    const bool do_copy = (tid < JT * 64);
    const int8_t* gsrc = B + (long)(nb + rr) * K + cc * 16;
    const uint32_t lds0 = (uint32_t)(uintptr_t)&ldsB[0][rr * 64 + cc * 16];
    const uint32_t lds1 = (uint32_t)(uintptr_t)&ldsB[1][rr * 64 + cc * 16];
    const int8_t* lb0 = &ldsB[0][0];
    const int8_t* lb1 = &ldsB[1][0];

    const int8_t* arow = A + (long)(m0 + l16) * K;

    if (do_copy) async_g2l_b128(gsrc, lds0);     // prologue: slab 0 -> buffer 0

    for (int k0 = 0; k0 < K; k0 += 128) {
        gemm_step<JT>(arow, k0,      K, lb0, do_copy, gsrc, lds1, half, l16, acc);
        gemm_step<JT>(arow, k0 + 64, K, lb1, do_copy, gsrc, lds0, half, l16, acc);
    }

    const float ws = __uint_as_float(*wmax_slot) * (1.f / 127.f);
    const float scale = prefac * ws;
#pragma unroll
    for (int j = 0; j < JT; ++j) {
        const int n0 = nb + j * 16;
#pragma unroll
        for (int r = 0; r < 8; ++r) {
            const int  row = m0 + half * 8 + r;   // C layout: VGPR r -> M = r (+8 hi half)
            const long idx = (long)row * N + n0 + l16;
            const float v = (float)acc[j][r] * scale;
            if (MODE == 0) ((float*)Out)[idx] = v;
            else           ((int8_t*)Out)[idx] = qa8(v, inv_qdiv);
        }
    }
}

// ---------------- depthwise causal conv(K=4) + SiLU + requant ----------------
__global__ void k_conv_silu(const int8_t* __restrict__ proj_q, const int8_t* __restrict__ cwq,
                            const unsigned* __restrict__ cw_slot, const float* __restrict__ conv_b,
                            int8_t* __restrict__ u_q) {
    const long idx = (long)blockIdx.x * 256 + threadIdx.x;   // 2^24 total
    const int d = (int)(idx & 4095);
    const int s = (int)((idx >> 12) & 2047);
    const int b = (int)(idx >> 23);
    int acc = 0;
#pragma unroll
    for (int k = 0; k < 4; ++k) {
        const int si = s - 3 + k;
        if (si >= 0)
            acc += (int)cwq[d * 4 + k] * (int)proj_q[(long)(b * 2048 + si) * 8192 + d];
    }
    const float cws = __uint_as_float(*cw_slot) * (1.f / 127.f);
    float v = (float)acc * (S_Z_f * cws) + conv_b[d];
    v = siluf_(v);
    u_q[idx] = qa8(v, 1.f / S_U_f);
}

// ---------------- per-token RMSNorm + quantize (dt / B / C slices) ----------------
__global__ void k_rmsnorm(const float* __restrict__ ssm_p,
                          const float* __restrict__ dt_ln_w,
                          const float* __restrict__ B_ln_w,
                          const float* __restrict__ C_ln_w,
                          int8_t* __restrict__ dt_q, float* __restrict__ BCc) {
    const int tok  = blockIdx.x;
    const int lane = threadIdx.x;
    const float* row = ssm_p + (long)tok * 160;

    float x[4], ss = 0.f;
#pragma unroll
    for (int i = 0; i < 4; ++i) { x[i] = row[lane * 4 + i]; ss += x[i] * x[i]; }
    for (int off = 16; off; off >>= 1) ss += __shfl_xor(ss, off, 32);
    const float rs = rsqrtf(ss * (1.f / 128.f) + EPS_f);
#pragma unroll
    for (int i = 0; i < 4; ++i)
        dt_q[(long)tok * 128 + lane * 4 + i] = qa8(dt_ln_w[lane * 4 + i] * x[i] * rs, 1.f / S_DTLN_f);

    // B (cols 128..143) in lanes 0..15, C (cols 144..159) in lanes 16..31
    const float v = row[128 + lane];
    float s2 = v * v;
    for (int off = 8; off; off >>= 1) s2 += __shfl_xor(s2, off, 16);   // within 16-groups
    const float rs2 = rsqrtf(s2 * (1.f / 16.f) + EPS_f);
    const float w  = (lane < 16) ? B_ln_w[lane] : C_ln_w[lane - 16];
    const float sq = (lane < 16) ? S_B_f : S_C_f;
    float q = rintf(w * v * rs2 / sq);
    q = fminf(fmaxf(q, -128.f), 127.f);
    BCc[(long)tok * 32 + lane] = q * sq;   // store de-quantized (B_q*S_B / C_q*S_C)
}

// ---------------- SSM scan (sequential over t, parallel over b,d) ----------------
__global__ void k_scan(const int8_t* __restrict__ dt8, const float* __restrict__ dt_bias,
                       const int8_t* __restrict__ u_q8, const float* __restrict__ BCc,
                       const float* __restrict__ A_log, const float* __restrict__ D_skip,
                       const int8_t* __restrict__ proj_q, int8_t* __restrict__ y_q8) {
    const int d = blockIdx.x * 256 + threadIdx.x;
    const int b = blockIdx.y;
    __shared__ float sBC[32];

    float Arow[16], h[16];
#pragma unroll
    for (int n = 0; n < 16; ++n) {
        Arow[n] = -__expf(A_log[(long)d * 16 + n]);
        h[n] = 0.f;
    }
    const float bias_d = dt_bias[d];
    const float Dsk    = D_skip[d];

    for (int t = 0; t < 2048; ++t) {
        const long tok = (long)b * 2048 + t;
        if (threadIdx.x < 32) sBC[threadIdx.x] = BCc[tok * 32 + threadIdx.x];
        __syncthreads();

        const float dtv = softplusf_((float)dt8[tok * 4096 + d] * S_DT_f + bias_d);
        const float u   = (float)u_q8[tok * 4096 + d] * S_U_f;
        const float du  = dtv * u;
        float y = 0.f;
#pragma unroll
        for (int n = 0; n < 16; ++n) {
            const float dA = __expf(dtv * Arow[n]);
            h[n] = h[n] * dA + du * sBC[n];
            y += h[n] * sBC[16 + n];
        }
        const float g  = (float)proj_q[tok * 8192 + 4096 + d] * S_Z_f;
        const float ys = (y + u * Dsk) * siluf_(g);
        y_q8[tok * 4096 + d] = qa8(ys, 1.f / S_Y_f);
        __syncthreads();
    }
}

// ---------------- host orchestration ----------------
extern "C" void kernel_launch(void* const* d_in, const int* in_sizes, int n_in,
                              void* d_out, int out_size, void* d_ws, size_t ws_size,
                              hipStream_t stream) {
    const float* hidden     = (const float*)d_in[0];
    const float* in_proj_w  = (const float*)d_in[1];
    const float* conv_w     = (const float*)d_in[2];
    const float* conv_b     = (const float*)d_in[3];
    const float* x_proj_w   = (const float*)d_in[4];
    const float* dt_ln_w    = (const float*)d_in[5];
    const float* B_ln_w     = (const float*)d_in[6];
    const float* C_ln_w     = (const float*)d_in[7];
    const float* dt_proj_w  = (const float*)d_in[8];
    const float* dt_bias    = (const float*)d_in[9];
    const float* A_log      = (const float*)d_in[10];
    const float* D_skip     = (const float*)d_in[11];
    const float* out_proj_w = (const float*)d_in[12];
    (void)in_sizes; (void)n_in; (void)out_size; (void)ws_size;

    const int  Bsz = 2, S = 2048, H = 2048, Din = 4096, Kc = 4;
    const long BS = (long)Bsz * S;   // 4096 tokens

    // workspace carve-out (256B aligned slabs), ~122 MB total
    uint8_t* p = (uint8_t*)d_ws;
    auto carve = [&](size_t bytes) -> void* {
        void* r = (void*)p;
        p += (bytes + 255) & ~(size_t)255;
        return r;
    };
    unsigned* scales = (unsigned*)carve(8 * sizeof(unsigned)); // abs-max slots 0..4
    int8_t* in_q   = (int8_t*)carve((size_t)2 * Din * H);      // (8192,2048)
    int8_t* xp_q   = (int8_t*)carve((size_t)160 * Din);        // (160,4096)
    int8_t* dtp_q  = (int8_t*)carve((size_t)Din * 128);        // (4096,128)
    int8_t* op_q   = (int8_t*)carve((size_t)H * Din);          // (2048,4096)
    int8_t* cw_q   = (int8_t*)carve((size_t)Din * Kc);
    int8_t* x_q    = (int8_t*)carve((size_t)BS * H);
    int8_t* proj_q = (int8_t*)carve((size_t)BS * 2 * Din);     // token-major (BS,8192)
    int8_t* u_q    = (int8_t*)carve((size_t)BS * Din);
    float*  ssm_p  = (float*)carve((size_t)BS * 160 * 4);
    int8_t* dt_q   = (int8_t*)carve((size_t)BS * 128);
    float*  BCc    = (float*)carve((size_t)BS * 32 * 4);
    int8_t* dt8    = (int8_t*)carve((size_t)BS * Din);
    int8_t* yq     = (int8_t*)carve((size_t)BS * Din);

    // 1) scale slots
    k_zero_u32<<<1, 8, 0, stream>>>(scales, 8);

    // 2) abs-max per weight
    auto amax = [&](const float* w, long n, int slot) {
        long blocks = (n + 255) / 256; if (blocks > 1024) blocks = 1024;
        k_absmax<<<(unsigned)blocks, 256, 0, stream>>>(w, n, scales + slot);
    };
    amax(in_proj_w,  (long)2 * Din * H, 0);
    amax(x_proj_w,   (long)160 * Din,   1);
    amax(dt_proj_w,  (long)Din * 128,   2);
    amax(out_proj_w, (long)H * Din,     3);
    amax(conv_w,     (long)Din * Kc,    4);

    // 3) quantize weights
    auto quantw = [&](const float* w, long n, int slot, int8_t* out) {
        k_quant_w<<<(unsigned)((n + 255) / 256), 256, 0, stream>>>(w, n, scales + slot, out);
    };
    quantw(in_proj_w,  (long)2 * Din * H, 0, in_q);
    quantw(x_proj_w,   (long)160 * Din,   1, xp_q);
    quantw(dt_proj_w,  (long)Din * 128,   2, dtp_q);
    quantw(out_proj_w, (long)H * Din,     3, op_q);
    quantw(conv_w,     (long)Din * Kc,    4, cw_q);

    // 4) quantize input activations
    k_quant_x<<<(unsigned)((BS * H + 255) / 256), 256, 0, stream>>>(
        hidden, BS * H, 1.f / S_IN_f, x_q);

    const unsigned gy = (unsigned)(BS / 128);   // M tiles (BS = 4096, exact)

    // 5) in_proj GEMM: proj_q = qa(x_q @ wq^T * S_IN*ws, S_Z)
    k_gemm_i8<1, 4><<<dim3(2 * Din / 64, gy), 256, 0, stream>>>(
        x_q, in_q, (int)BS, 2 * Din, H, 0, proj_q, scales + 0, S_IN_f, 1.f / S_Z_f);

    // 6) depthwise causal conv + SiLU + requant -> u_q (token-major)
    k_conv_silu<<<(unsigned)((BS * Din) / 256), 256, 0, stream>>>(
        proj_q, cw_q, scales + 4, conv_b, u_q);

    // 7) x_proj GEMM: ssm_p = u_q @ xwq^T * S_U*xws  (float out, N=160 = 2x64 + 32 tail)
    k_gemm_i8<0, 4><<<dim3(2, gy), 256, 0, stream>>>(
        u_q, xp_q, (int)BS, 160, Din, 0, ssm_p, scales + 1, S_U_f, 0.f);
    k_gemm_i8<0, 2><<<dim3(1, gy), 256, 0, stream>>>(
        u_q, xp_q, (int)BS, 160, Din, 128, ssm_p, scales + 1, S_U_f, 0.f);

    // 8) RMSNorm+quant of dt / B / C slices
    k_rmsnorm<<<(unsigned)BS, 32, 0, stream>>>(ssm_p, dt_ln_w, B_ln_w, C_ln_w, dt_q, BCc);

    // 9) dt_proj GEMM: dt8 = qa(dt_q @ dwq^T * S_DTLN*dws, S_DT)
    k_gemm_i8<1, 4><<<dim3(Din / 64, gy), 256, 0, stream>>>(
        dt_q, dtp_q, (int)BS, Din, 128, 0, dt8, scales + 2, S_DTLN_f, 1.f / S_DT_f);

    // 10) SSM scan (fuses softplus(dt), D-skip, SiLU(gate), qa(., S_Y))
    k_scan<<<dim3(Din / 256, Bsz), 256, 0, stream>>>(
        dt8, dt_bias, u_q, BCc, A_log, D_skip, proj_q, yq);

    // 11) out_proj GEMM: out = y_q @ owq^T * S_Y*ows (float)
    k_gemm_i8<0, 4><<<dim3(H / 64, gy), 256, 0, stream>>>(
        yq, op_q, (int)BS, H, Din, 0, d_out, scales + 3, S_Y_f, 0.f);
}
